// AttentionBlock_19885698580995
// MI455X (gfx1250) — matrix operand order repaired and checked by reading, not compiled, not run on hardware
//
#include <hip/hip_runtime.h>
#include <hip/hip_bf16.h>

typedef __attribute__((ext_vector_type(16))) _Float16 v16h;
typedef __attribute__((ext_vector_type(8)))  _Float16 v8h;
typedef __attribute__((ext_vector_type(8)))  float    v8f;
typedef __attribute__((ext_vector_type(4)))  unsigned int u32x4;
typedef __attribute__((ext_vector_type(8)))  unsigned int u32x8;

#define SEQ 128
#define CH  64
#define BN_EPS 1e-3f

// ---- LDS pool layout (bytes) ----
// Persistent regions:
#define LDS_WT   0        // 32768 : _Float16 sWT[4][64][64] (transposed [n][k])
#define LDS_XN   32768    // 32768 : float    sXn[128][64]
#define LDS_Q    65536    // 16384 : _Float16 sQ[128][64]
#define LDS_K    81920    // 16384 : _Float16 sK[128][64]
#define LDS_VT   98304    // 16384 : _Float16 sVT[64][128]
#define LDS_P    114688   // 32768 : _Float16 sP[8][16][128]
#define LDS_O    147456   // 16384 : _Float16 sO[128][64]
#define LDS_PAR  163840   // 1536  : bias[4][64], scale[64], shift[64]
#define LDS_TOTAL 165376
// Transient staging (consumed before Q/K/VT/P/O are first written):
#define LDS_STGW LDS_Q           // 65536 : raw f32 Wq,Wk,Wv,Wo (16KB each)
#define LDS_STGX (LDS_Q + 65536) // 32768 : raw f32 x tile  (ends at LDS_PAR)

__device__ __forceinline__ v16h pack16(v8h lo, v8h hi) {
  v16h a;
#pragma unroll
  for (int e = 0; e < 8; ++e) { a[e] = lo[e]; a[e + 8] = hi[e]; }
  return a;
}

// A fragment (16x32 f16, MxK) from row-major f16 LDS tile.
__device__ __forceinline__ v16h loadA_h(const _Float16* S, int ld, int row0,
                                        int k0, int lane) {
  const int r = lane & 15, g = lane >> 4;
  const _Float16* p = S + (row0 + r) * ld + k0 + g * 8;
  v8h lo = *(const v8h*)(p);
  v8h hi = *(const v8h*)(p + 16);
  return pack16(lo, hi);
}

// A fragment from row-major f32 LDS tile (convert on load).
__device__ __forceinline__ v16h loadA_f32(const float* S, int ld, int row0,
                                          int k0, int lane) {
  const int r = lane & 15, g = lane >> 4;
  const float* p = S + (row0 + r) * ld + k0 + g * 8;
  v16h a;
#pragma unroll
  for (int e = 0; e < 8; ++e) {
    a[e]     = (_Float16)p[e];
    a[e + 8] = (_Float16)p[e + 16];
  }
  return a;
}

// B fragment (32x16 f16, KxN) from transposed storage BT[n][k].
__device__ __forceinline__ v16h loadB_T(const _Float16* S, int ld, int n0,
                                        int k0, int lane) {
  const int n = lane & 15, g = lane >> 4;
  const _Float16* p = S + (n0 + n) * ld + k0 + g * 8;
  v8h lo = *(const v8h*)(p);
  v8h hi = *(const v8h*)(p + 16);
  return pack16(lo, hi);
}

__device__ __forceinline__ v8f wmma16(v16h a, v16h b, v8f c) {
  return __builtin_amdgcn_wmma_f32_16x16x32_f16(false, a, false, b, (short)0,
                                                c, false, false);
}

__device__ __forceinline__ float redmax16(float v) {
#pragma unroll
  for (int m = 8; m >= 1; m >>= 1) v = fmaxf(v, __shfl_xor(v, m, 32));
  return v;
}
__device__ __forceinline__ float redsum16(float v) {
#pragma unroll
  for (int m = 8; m >= 1; m >>= 1) v += __shfl_xor(v, m, 32);
  return v;
}

// Tensor Data Mover: 1-D contiguous copy of `nelem` f32 from global to LDS.
// D# per CDNA5 ISA. Issued once per executing wave
// (EXEC ignored by TDM); tracked by TENSORcnt.
__device__ __forceinline__ void tdm_load_1d_f32(unsigned int lds_addr,
                                                const float* gsrc,
                                                unsigned int nelem) {
  const unsigned long long ga = (unsigned long long)(uintptr_t)gsrc;
  u32x4 g0;
  g0[0] = 1u;                                   // count=1, is_restore=0
  g0[1] = lds_addr;                             // lds_addr [63:32]
  g0[2] = (unsigned int)ga;                     // global_addr [95:64]
  g0[3] = (unsigned int)(ga >> 32) | (2u << 30); // addr[56:32] | type=2
  u32x8 g1;
  g1[0] = 2u << 16;                // data_size=2 (4B), mask=0, no pad/iter
  g1[1] = nelem << 16;             // tensor_dim0[15:0] in [31:16]
  g1[2] = (nelem >> 16) | (1u << 16); // tensor_dim0[31:16] | tensor_dim1=1
  g1[3] = nelem << 16;             // tile_dim0 in [31:16]
  g1[4] = 1u;                      // tile_dim1=1, tile_dim2=0
  g1[5] = nelem;                   // tensor_dim0_stride[31:0]
  g1[6] = 0u;
  g1[7] = 0u;
  asm volatile("tensor_load_to_lds %0, %1" :: "s"(g0), "s"(g1) : "memory");
}

__global__ __launch_bounds__(256, 1)
void attn_block_kernel(const float* __restrict__ x,
                       const float* __restrict__ gamma,
                       const float* __restrict__ beta,
                       const float* __restrict__ mmean,
                       const float* __restrict__ mvar,
                       const float* __restrict__ Wq, const float* __restrict__ bq,
                       const float* __restrict__ Wk, const float* __restrict__ bk,
                       const float* __restrict__ Wv, const float* __restrict__ bv,
                       const float* __restrict__ Wo, const float* __restrict__ bo,
                       float* __restrict__ out) {
  __shared__ __align__(16) unsigned char smem[LDS_TOTAL];

  _Float16* sWT   = (_Float16*)(smem + LDS_WT);
  float*    sXn   = (float*)(smem + LDS_XN);
  _Float16* sQ    = (_Float16*)(smem + LDS_Q);
  _Float16* sK    = (_Float16*)(smem + LDS_K);
  _Float16* sVT   = (_Float16*)(smem + LDS_VT);
  _Float16* sP    = (_Float16*)(smem + LDS_P);
  _Float16* sO    = (_Float16*)(smem + LDS_O);
  float*    sBias = (float*)(smem + LDS_PAR);          // 4*64
  float*    sScale= (float*)(smem + LDS_PAR + 1024);   // 64
  float*    sShift= (float*)(smem + LDS_PAR + 1280);   // 64
  const float* sWraw = (const float*)(smem + LDS_STGW);

  const int tid  = threadIdx.x;
  const int lane = tid & 31;
  const int wave = tid >> 5;
  const int ln   = lane & 15;
  const int g    = lane >> 4;
  const size_t bh = blockIdx.x;
  const float* xin = x + bh * (size_t)(SEQ * CH);
  float* outp = out + bh * (size_t)(SEQ * CH);

  // ---- Issue TDM DMAs (wave 0): 4 weight tiles + x tile -> LDS staging ----
  if (tid < 32) {
    const unsigned int base = (unsigned int)(uintptr_t)(void*)smem;
    tdm_load_1d_f32(base + LDS_STGW + 0 * 16384, Wq, CH * CH);
    tdm_load_1d_f32(base + LDS_STGW + 1 * 16384, Wk, CH * CH);
    tdm_load_1d_f32(base + LDS_STGW + 2 * 16384, Wv, CH * CH);
    tdm_load_1d_f32(base + LDS_STGW + 3 * 16384, Wo, CH * CH);
    tdm_load_1d_f32(base + LDS_STGX, xin, SEQ * CH);
  }

  // ---- Meanwhile: BN params + biases (tiny scalar loads, hit L2) ----
  if (tid >= 32 && tid < 96) {
    const int c = tid - 32;
    const float sc = gamma[c] * rsqrtf(mvar[c] + BN_EPS);
    sScale[c] = sc;
    sShift[c] = beta[c] - mmean[c] * sc;
    sBias[0 * CH + c] = bq[c];
    sBias[1 * CH + c] = bk[c];
    sBias[2 * CH + c] = bv[c];
    sBias[3 * CH + c] = bo[c];
  }

  if (tid < 32) __builtin_amdgcn_s_wait_tensorcnt(0);
  __syncthreads();  // staged raw data + params visible to all waves

  // ---- Convert weights: raw f32 [k][n] -> f16 transposed sWT[n][k] ----
  for (int i = tid; i < 4 * CH * CH; i += 256) {
    const int w = i >> 12, idx = i & 4095;
    const int k = idx >> 6, n = idx & 63;
    sWT[w * (CH * CH) + n * CH + k] = (_Float16)sWraw[i];
  }

  // ---- BatchNorm: raw x (LDS) -> xn f32 (LDS), float4 vectorized ----
  {
    const float4* sXraw4 = (const float4*)(smem + LDS_STGX);
    float4* sXn4 = (float4*)sXn;
    for (int i = tid; i < (SEQ * CH) / 4; i += 256) {
      float4 v = sXraw4[i];
      const int c = (i * 4) & (CH - 1);
      v.x = v.x * sScale[c + 0] + sShift[c + 0];
      v.y = v.y * sScale[c + 1] + sShift[c + 1];
      v.z = v.z * sScale[c + 2] + sShift[c + 2];
      v.w = v.w * sScale[c + 3] + sShift[c + 3];
      sXn4[i] = v;
    }
  }
  __syncthreads();  // staging regions now dead; Q/K/VT/P/O may be written

  const int m0 = wave * 16;  // this wave's 16 query rows

  // ---- QKV projections: [16x64] = xn[16x64] @ W[64x64] + b ----
  v16h Ax[2];
#pragma unroll
  for (int kt = 0; kt < 2; ++kt) Ax[kt] = loadA_f32(sXn, CH, m0, kt * 32, lane);

#pragma unroll
  for (int mm = 0; mm < 3; ++mm) {
#pragma unroll
    for (int nt = 0; nt < 4; ++nt) {
      v8f acc = {};
#pragma unroll
      for (int kt = 0; kt < 2; ++kt)
        acc = wmma16(Ax[kt],
                     loadB_T(sWT + mm * (CH * CH), CH, nt * 16, kt * 32, lane),
                     acc);
      const float bias = sBias[mm * CH + nt * 16 + ln];
#pragma unroll
      for (int i = 0; i < 8; ++i) {
        const float vv = acc[i] + bias;
        const int row = m0 + g * 8 + i;
        const int col = nt * 16 + ln;
        if (mm == 0)      sQ[row * CH + col]   = (_Float16)vv;
        else if (mm == 1) sK[row * CH + col]   = (_Float16)vv;
        else              sVT[col * SEQ + row] = (_Float16)vv;  // transposed
      }
    }
  }
  __syncthreads();  // all waves' K,V visible

  // ---- Scores: S[16x128] = Q[16x64] @ K^T, kept in registers ----
  v16h Aq[2];
#pragma unroll
  for (int kt = 0; kt < 2; ++kt) Aq[kt] = loadA_h(sQ, CH, m0, kt * 32, lane);

  v8f Sc[8];
#pragma unroll
  for (int nt = 0; nt < 8; ++nt) {
    v8f acc = {};
#pragma unroll
    for (int kt = 0; kt < 2; ++kt)
      acc = wmma16(Aq[kt], loadB_T(sK, CH, nt * 16, kt * 32, lane), acc);
    Sc[nt] = acc;
  }

  // ---- Softmax over 128 keys (row = g*8+i; cols across 16 lanes x 8 tiles) ----
  _Float16* myP = sP + wave * 16 * SEQ;
#pragma unroll
  for (int i = 0; i < 8; ++i) {
    float m = -3.4e38f;
#pragma unroll
    for (int nt = 0; nt < 8; ++nt) m = fmaxf(m, Sc[nt][i]);
    m = redmax16(m);
    float s = 0.f;
#pragma unroll
    for (int nt = 0; nt < 8; ++nt) {
      const float e = __expf((Sc[nt][i] - m) * 0.125f);  // d^-0.5 = 1/8
      Sc[nt][i] = e;
      s += e;
    }
    const float inv = 1.0f / redsum16(s);
#pragma unroll
    for (int nt = 0; nt < 8; ++nt)
      myP[(g * 8 + i) * SEQ + nt * 16 + ln] = (_Float16)(Sc[nt][i] * inv);
  }
  // myP private to this wave; same-wave LDS ops are in-order.

  // ---- out = P[16x128] @ V[128x64]  (B from transposed VT) ----
  v16h Ap[4];
#pragma unroll
  for (int kt = 0; kt < 4; ++kt) Ap[kt] = loadA_h(myP, SEQ, 0, kt * 32, lane);
#pragma unroll
  for (int nt = 0; nt < 4; ++nt) {
    v8f acc = {};
#pragma unroll
    for (int kt = 0; kt < 4; ++kt)
      acc = wmma16(Ap[kt], loadB_T(sVT, SEQ, nt * 16, kt * 32, lane), acc);
#pragma unroll
    for (int i = 0; i < 8; ++i)
      sO[(m0 + g * 8 + i) * CH + nt * 16 + ln] = (_Float16)acc[i];
  }

  // ---- Output projection + bias + residual(xn), f32 to global ----
  v16h Ao[2];
#pragma unroll
  for (int kt = 0; kt < 2; ++kt) Ao[kt] = loadA_h(sO, CH, m0, kt * 32, lane);
#pragma unroll
  for (int nt = 0; nt < 4; ++nt) {
    v8f acc = {};
#pragma unroll
    for (int kt = 0; kt < 2; ++kt)
      acc = wmma16(Ao[kt],
                   loadB_T(sWT + 3 * (CH * CH), CH, nt * 16, kt * 32, lane),
                   acc);
    const float bias = sBias[3 * CH + nt * 16 + ln];
#pragma unroll
    for (int i = 0; i < 8; ++i) {
      const int row = m0 + g * 8 + i;
      const int col = nt * 16 + ln;
      outp[row * CH + col] = acc[i] + bias + sXn[row * CH + col];
    }
  }
}

extern "C" void kernel_launch(void* const* d_in, const int* in_sizes, int n_in,
                              void* d_out, int out_size, void* d_ws,
                              size_t ws_size, hipStream_t stream) {
  const float* x     = (const float*)d_in[0];
  const float* gamma = (const float*)d_in[1];
  const float* beta  = (const float*)d_in[2];
  const float* mmean = (const float*)d_in[3];
  const float* mvar  = (const float*)d_in[4];
  const float* Wq = (const float*)d_in[5];  const float* bq = (const float*)d_in[6];
  const float* Wk = (const float*)d_in[7];  const float* bk = (const float*)d_in[8];
  const float* Wv = (const float*)d_in[9];  const float* bv = (const float*)d_in[10];
  const float* Wo = (const float*)d_in[11]; const float* bo = (const float*)d_in[12];
  float* out = (float*)d_out;

  const int nBH = 64 * 128;  // B*H independent attention problems
  attn_block_kernel<<<dim3(nBH), dim3(256), 0, stream>>>(
      x, gamma, beta, mmean, mvar, Wq, bq, Wk, bk, Wv, bv, Wo, bo, out);
}
